// GCNnet_14800457302512
// MI455X (gfx1250) — matrix operand order repaired and checked by reading, not compiled, not run on hardware
//
#include <hip/hip_runtime.h>
#include <math.h>

// ---------------------------------------------------------------------------
// GCN (3x GCNConv + mean-pool + MLP) for MI455X / gfx1250, wave32.
// GEMMs: operands pre-packed into the ISA 7.12.2 fragment layout (f16, zero
// padded), so the WMMA inner loop is 2x b128 loads + v_wmma_f32_16x16x32_f16
// per k-block with no branches. Edge aggregation is float4-gather +
// global_atomic_add_f32 scatter (working set fits the 192MB L2).
// ---------------------------------------------------------------------------

typedef __attribute__((ext_vector_type(16))) _Float16 v16h;
typedef __attribute__((ext_vector_type(8)))  float    v8f;

#define WPB 4  // waves per block in the GEMM kernel

// ---------------- simple elementwise kernels ----------------

__global__ void k_fill(float* __restrict__ p, int n, float v) {
  int i = blockIdx.x * blockDim.x + threadIdx.x;
  if (i < n) p[i] = v;
}

__global__ void k_count_deg(const int* __restrict__ dst, float* __restrict__ deg, int E) {
  int i = blockIdx.x * blockDim.x + threadIdx.x;
  if (i < E) atomicAdd(&deg[dst[i]], 1.0f);
}

__global__ void k_rsqrt_inplace(float* __restrict__ p, int n) {
  int i = blockIdx.x * blockDim.x + threadIdx.x;
  if (i < n) p[i] = rsqrtf(p[i]);
}

// agg[n][f] = hlin[n][f] * dis[n]^2   (self-loop contribution, also init)
template <int F>
__global__ void k_selfloop_init(const float* __restrict__ hlin, const float* __restrict__ dis,
                                float* __restrict__ agg, int total) {
  int i = blockIdx.x * blockDim.x + threadIdx.x;
  if (i < total) {
    float d = dis[i / F];
    agg[i] = hlin[i] * d * d;
  }
}

// agg[dst] += hlin[src] * dis[src]*dis[dst]; one float4 chunk per thread,
// edge-major mapping so a wave touches ~contiguous source-row bytes.
template <int F4>
__global__ void k_edge_scatter(const int* __restrict__ src, const int* __restrict__ dst,
                               const float* __restrict__ dis, const float* __restrict__ hlin,
                               float* __restrict__ agg, int total) {
  int i = blockIdx.x * blockDim.x + threadIdx.x;
  if (i >= total) return;
  int e = i / F4;
  int c = i - e * F4;
  int s = src[e];
  int d = dst[e];
  float w = dis[s] * dis[d];
  float4 v = ((const float4*)hlin)[(size_t)s * F4 + c];
  float* o = agg + ((size_t)d * F4 + c) * 4;
  atomicAdd(o + 0, v.x * w);
  atomicAdd(o + 1, v.y * w);
  atomicAdd(o + 2, v.z * w);
  atomicAdd(o + 3, v.w * w);
}

template <int F>
__global__ void k_bias_relu(float* __restrict__ h, const float* __restrict__ b, int total) {
  int i = blockIdx.x * blockDim.x + threadIdx.x;
  if (i < total) {
    float v = h[i] + b[i % F];
    h[i] = v > 0.0f ? v : 0.0f;
  }
}

__global__ void k_batch_count(const int* __restrict__ batch, float* __restrict__ counts, int n) {
  int i = blockIdx.x * blockDim.x + threadIdx.x;
  if (i < n) atomicAdd(&counts[batch[i]], 1.0f);
}

template <int F>
__global__ void k_pool_acc(const float* __restrict__ h, const int* __restrict__ batch,
                           float* __restrict__ pooled, int total) {
  int i = blockIdx.x * blockDim.x + threadIdx.x;
  if (i < total) {
    int n = i / F;
    int f = i - n * F;
    atomicAdd(&pooled[batch[n] * F + f], h[i]);
  }
}

template <int F>
__global__ void k_pool_div(float* __restrict__ pooled, const float* __restrict__ counts,
                           int total) {
  int i = blockIdx.x * blockDim.x + threadIdx.x;
  if (i < total) pooled[i] /= fmaxf(counts[i / F], 1.0f);
}

// z = z1 @ fW2 + fb2 ; softmax over a singleton axis is identically 1.0
__global__ void k_fc2_softmax(const float* __restrict__ z1, const float* __restrict__ fW2,
                              const float* __restrict__ fb2, float* __restrict__ out,
                              int G, int K) {
  int g = blockIdx.x * blockDim.x + threadIdx.x;
  if (g < G) {
    float acc = fb2[0];
    for (int k = 0; k < K; ++k) acc += z1[g * K + k] * fW2[k];
    out[g] = expf(acc - acc);  // softmax(singleton) == 1.0
  }
}

// ---------------- WMMA operand packing ----------------
// Fragment-major layouts (ISA 7.12.2, 16-bit 16x32 A / 32x16 B):
//   Ap[(mt*KB + kb)*256 + lane*8 + v]  (uint = packed half2)
//     lane: m = lane&15; dword v holds K = kb*32 + (v>=4?16:0) + (lane>=16?8:0)
//                                        + 2*(v&3), +1
//   Bp[(nt*KB + kb)*256 + lane*8 + v]
//     lane: n = lane&15; dword v holds K = same k order as Ap
// Out-of-range elements are zero-filled, so the GEMM loop is branch-free.

union h2u { _Float16 h[2]; unsigned u; };

__global__ void k_pack_a(const float* __restrict__ A, unsigned* __restrict__ Ap,
                         int M, int K) {
  int tid = threadIdx.x;           // 0..255
  int v = tid & 7, lane = tid >> 3;
  int kb = blockIdx.x, mt = blockIdx.y;
  int half = lane >> 4, l16 = lane & 15;
  int row = mt * 16 + l16;
  int ka = kb * 32 + ((v & 4) << 2) + (half << 3) + ((v & 3) << 1);
  float a0 = (row < M && ka     < K) ? A[(size_t)row * K + ka]     : 0.0f;
  float a1 = (row < M && ka + 1 < K) ? A[(size_t)row * K + ka + 1] : 0.0f;
  h2u p;
  p.h[0] = (_Float16)a0;
  p.h[1] = (_Float16)a1;
  Ap[((size_t)mt * gridDim.x + kb) * 256 + tid] = p.u;
}

__global__ void k_pack_b(const float* __restrict__ B, unsigned* __restrict__ Bp,
                         int N, int K) {
  int tid = threadIdx.x;
  int v = tid & 7, lane = tid >> 3;
  int kb = blockIdx.x, nt = blockIdx.y;
  int half = lane >> 4, l16 = lane & 15;
  int col = nt * 16 + l16;
  int kw = kb * 32 + ((v & 4) << 2) + (half << 3) + ((v & 3) << 1);
  float b0 = (col < N && kw     < K) ? B[(size_t)kw * N + col]       : 0.0f;
  float b1 = (col < N && kw + 1 < K) ? B[(size_t)(kw + 1) * N + col] : 0.0f;
  h2u p;
  p.h[0] = (_Float16)b0;
  p.h[1] = (_Float16)b1;
  Bp[((size_t)nt * gridDim.x + kb) * 256 + tid] = p.u;
}

// ---------------- WMMA GEMM on packed operands ----------------
// One wave computes one 16x16 C tile; inner loop: 32B A load + 32B B load +
// v_wmma_f32_16x16x32_f16, no branches, no per-element waits.
__global__ void k_wmma_gemm(const v16h* __restrict__ Ap, const v16h* __restrict__ Bp,
                            float* __restrict__ C, int M, int N, int KB) {
  const int lane = threadIdx.x & 31;
  const int wv   = threadIdx.x >> 5;
  const int mt   = blockIdx.x * WPB + wv;
  const int nt   = blockIdx.y;
  if (mt * 16 >= M) return;  // wave-uniform: EXEC stays all-1s for WMMA

  const v16h* ap = Ap + (size_t)mt * KB * 32 + lane;
  const v16h* bp = Bp + (size_t)nt * KB * 32 + lane;
  v8f acc = {};
  for (int kb = 0; kb < KB; ++kb) {
    v16h a = ap[0];
    v16h b = bp[0];
    acc = __builtin_amdgcn_wmma_f32_16x16x32_f16(false, a, false, b,
                                                 (short)0, acc, false, false);
    ap += 32;
    bp += 32;
  }
  const int half = lane >> 4, l16 = lane & 15;
  const int col = nt * 16 + l16;
  if (col < N) {
#pragma unroll
    for (int r = 0; r < 8; ++r) {
      int mrow = mt * 16 + r + (half << 3);
      if (mrow < M) C[(size_t)mrow * N + col] = acc[r];
    }
  }
}

// ---------------------------------------------------------------------------

static inline int ceil_div(int a, int b) { return (a + b - 1) / b; }

extern "C" void kernel_launch(void* const* d_in, const int* in_sizes, int n_in,
                              void* d_out, int out_size, void* d_ws, size_t ws_size,
                              hipStream_t stream) {
  (void)n_in; (void)ws_size;
  const float* x     = (const float*)d_in[0];
  const int*   ei    = (const int*)d_in[1];
  const int*   batch = (const int*)d_in[2];
  const float* W1 = (const float*)d_in[3];  const float* b1 = (const float*)d_in[4];
  const float* W2 = (const float*)d_in[5];  const float* b2 = (const float*)d_in[6];
  const float* W3 = (const float*)d_in[7];  const float* b3 = (const float*)d_in[8];
  const float* fW1 = (const float*)d_in[9];  const float* fb1 = (const float*)d_in[10];
  const float* fW2 = (const float*)d_in[11]; const float* fb2 = (const float*)d_in[12];

  const int F0 = 40, F1 = 40, F2 = 80, F3 = 160;
  const int Nn = in_sizes[0] / F0;     // 100000 nodes
  const int E  = in_sizes[1] / 2;      // 1600000 edges
  const int G  = out_size;             // 128 graphs
  const int* src = ei;
  const int* dst = ei + E;

  // workspace layout (~161 MB; harness-provided d_ws assumed sufficient)
  auto alignup = [](size_t v) { return (v + 255) & ~(size_t)255; };
  char* ws = (char*)d_ws;
  size_t off = 0;
  float*    dis    = (float*)(ws + off);    off += alignup((size_t)Nn * 4);
  float*    bufA   = (float*)(ws + off);    off += alignup((size_t)Nn * F3 * 4);
  float*    bufB   = (float*)(ws + off);    off += alignup((size_t)Nn * F3 * 4);
  unsigned* apack  = (unsigned*)(ws + off); off += alignup((size_t)6250 * 5 * 256 * 4);
  unsigned* bpack  = (unsigned*)(ws + off); off += alignup((size_t)16 * 5 * 256 * 4);
  float*    pooled = (float*)(ws + off);    off += alignup((size_t)G * F3 * 4);
  float*    z1     = (float*)(ws + off);    off += alignup((size_t)G * 128 * 4);
  float*    counts = (float*)(ws + off);    off += alignup((size_t)G * 4);

  const int TPB = 256;

  auto gemm = [&](const float* A, const float* B, float* C, int M, int K, int N) {
    int KB = ceil_div(K, 32), MT = ceil_div(M, 16), NT = ceil_div(N, 16);
    k_pack_a<<<dim3(KB, MT), 256, 0, stream>>>(A, apack, M, K);
    k_pack_b<<<dim3(KB, NT), 256, 0, stream>>>(B, bpack, N, K);
    dim3 grid(ceil_div(MT, WPB), NT);
    k_wmma_gemm<<<grid, 32 * WPB, 0, stream>>>((const v16h*)apack, (const v16h*)bpack,
                                               C, M, N, KB);
  };

  auto conv_agg = [&](const float* hlin, float* agg, const float* bias, int F) {
    int tot = Nn * F;
    int blocks = ceil_div(tot, TPB);
    if (F == 40)  k_selfloop_init<40><<<blocks, TPB, 0, stream>>>(hlin, dis, agg, tot);
    if (F == 80)  k_selfloop_init<80><<<blocks, TPB, 0, stream>>>(hlin, dis, agg, tot);
    if (F == 160) k_selfloop_init<160><<<blocks, TPB, 0, stream>>>(hlin, dis, agg, tot);
    int F4 = F / 4;
    int etot = E * F4;
    int eblocks = ceil_div(etot, TPB);
    if (F4 == 10) k_edge_scatter<10><<<eblocks, TPB, 0, stream>>>(src, dst, dis, hlin, agg, etot);
    if (F4 == 20) k_edge_scatter<20><<<eblocks, TPB, 0, stream>>>(src, dst, dis, hlin, agg, etot);
    if (F4 == 40) k_edge_scatter<40><<<eblocks, TPB, 0, stream>>>(src, dst, dis, hlin, agg, etot);
    if (F == 40)  k_bias_relu<40><<<blocks, TPB, 0, stream>>>(agg, bias, tot);
    if (F == 80)  k_bias_relu<80><<<blocks, TPB, 0, stream>>>(agg, bias, tot);
    if (F == 160) k_bias_relu<160><<<blocks, TPB, 0, stream>>>(agg, bias, tot);
  };

  // degree -> dis = rsqrt(1 + indeg)
  k_fill<<<ceil_div(Nn, TPB), TPB, 0, stream>>>(dis, Nn, 1.0f);
  k_count_deg<<<ceil_div(E, TPB), TPB, 0, stream>>>(dst, dis, E);
  k_rsqrt_inplace<<<ceil_div(Nn, TPB), TPB, 0, stream>>>(dis, Nn);

  // conv1: x(Nx40) @ W1(40x40) -> bufB; aggregate -> bufA (h1)
  gemm(x, W1, bufB, Nn, F0, F1);
  conv_agg(bufB, bufA, b1, F1);
  // conv2: h1 @ W2(40x80) -> bufB; aggregate -> bufA (h2)
  gemm(bufA, W2, bufB, Nn, F1, F2);
  conv_agg(bufB, bufA, b2, F2);
  // conv3: h2 @ W3(80x160) -> bufB; aggregate -> bufA (h3)
  gemm(bufA, W3, bufB, Nn, F2, F3);
  conv_agg(bufB, bufA, b3, F3);

  // global mean pool
  k_fill<<<ceil_div(G * F3, TPB), TPB, 0, stream>>>(pooled, G * F3, 0.0f);
  k_fill<<<1, TPB, 0, stream>>>(counts, G, 0.0f);
  k_batch_count<<<ceil_div(Nn, TPB), TPB, 0, stream>>>(batch, counts, Nn);
  k_pool_acc<160><<<ceil_div(Nn * F3, TPB), TPB, 0, stream>>>(bufA, batch, pooled, Nn * F3);
  k_pool_div<160><<<ceil_div(G * F3, TPB), TPB, 0, stream>>>(pooled, counts, G * F3);

  // FC1: pooled(128x160) @ fW1(160x128) -> z1, bias+relu
  gemm(pooled, fW1, z1, G, F3, 128);
  k_bias_relu<128><<<ceil_div(G * 128, TPB), TPB, 0, stream>>>(z1, fb1, G * 128);

  // FC2 + singleton softmax -> d_out
  k_fc2_softmax<<<ceil_div(G, TPB), TPB, 0, stream>>>(z1, fW2, fb2, (float*)d_out, G, 128);
}
